// MELMask_67035849556465
// MI455X (gfx1250) — matrix operand order repaired and checked by reading, not compiled, not run on hardware
//
#include <hip/hip_runtime.h>
#include <hip/hip_bf16.h>

typedef __attribute__((ext_vector_type(16))) _Float16 v16h;
typedef __attribute__((ext_vector_type(8)))  _Float16 v8h;
typedef __attribute__((ext_vector_type(8)))  float    v8f;

#define CDIM   640
#define HW     25
#define NWAY   5
#define KSHOT  5
#define QCNT   75
#define BATCH  32
#define MS     125     // n_way * hw
#define JPAD   128
#define GAMMA1 20.0f
#define GAMMA2 10.0f
#define KF     0.999f

// ---------------------------------------------------------------------------
// Kernel 1: support mean over k_shot + L2-norm over c, emit f16 Sh[b][128][640]
// Row j = n*25+p holds sup_n[b,n,:,p] contiguous in c (B-fragment friendly).
// Rows 125..127 zero padded.
// ---------------------------------------------------------------------------
__global__ void k_prep_support(const float* __restrict__ sup, _Float16* __restrict__ Sh) {
  const int blk = blockIdx.x;
  const int b = blk / NWAY, n = blk % NWAY;
  const int tid = threadIdx.x;

  __shared__ float mbuf[CDIM * HW];   // mean feature map [c][p]
  __shared__ float ssq[32];
  __shared__ float scl[32];

  if (tid < 32) ssq[tid] = 0.0f;
  __syncthreads();

  const float* base = sup + (size_t)(b * NWAY * KSHOT + n * KSHOT) * CDIM * HW;
  for (int idx = tid; idx < CDIM * HW; idx += blockDim.x) {
    float acc = 0.0f;
#pragma unroll
    for (int ks = 0; ks < KSHOT; ++ks) acc += base[(size_t)ks * CDIM * HW + idx];
    acc *= (1.0f / KSHOT);
    mbuf[idx] = acc;
    atomicAdd(&ssq[idx % HW], acc * acc);
  }
  __syncthreads();
  if (tid < 32) scl[tid] = rsqrtf(ssq[tid] + 1e-12f);
  __syncthreads();

  _Float16* out = Sh + ((size_t)b * JPAD + n * HW) * CDIM;
  for (int idx = tid; idx < CDIM * HW; idx += blockDim.x) {
    int c = idx / HW, p = idx % HW;
    out[(size_t)p * CDIM + c] = (_Float16)(mbuf[idx] * scl[p]);
  }
  if (n == 0) { // zero pad rows 125..127 of this batch
    _Float16* pad = Sh + ((size_t)b * JPAD + MS) * CDIM;
    for (int idx = tid; idx < (JPAD - MS) * CDIM; idx += blockDim.x)
      pad[idx] = (_Float16)0.0f;
  }
}

__device__ __forceinline__ float wave_max(float v) {
#pragma unroll
  for (int o = 16; o > 0; o >>= 1) v = fmaxf(v, __shfl_xor(v, o, 32));
  return v;
}
__device__ __forceinline__ float wave_sum(float v) {
#pragma unroll
  for (int o = 16; o > 0; o >>= 1) v += __shfl_xor(v, o, 32);
  return v;
}

// ---------------------------------------------------------------------------
// Kernel 2: one workgroup per (b,q). Fused: qry norm -> WMMA S -> softmaxes ->
// 25x25 Schur-complement Katz solve -> normalized outputs.
// ---------------------------------------------------------------------------
__global__ void __launch_bounds__(256)
k_katz(const float* __restrict__ qry, const _Float16* __restrict__ Sh,
       float* __restrict__ out_q, float* __restrict__ out_s) {
  const int inst = blockIdx.x;              // 0..2399
  const int b = inst / QCNT, qi = inst % QCNT;
  const int tid = threadIdx.x;
  const int lane = tid & 31, wave = tid >> 5;

  // ---- manually overlaid shared memory (< 64KB static) ----
  __shared__ __align__(16) unsigned char smem[59392];
  _Float16* Qh  = (_Float16*)smem;                 // [32][640] f16, stages 1-3
  float* Tsq    = (float*)(smem);                  // [25][128], stage 4+ (reuses Qh)
  float* Tqs    = (float*)(smem + 12800);          // [25][128]
  float* Mx     = (float*)(smem + 25600);          // [25][32]  (col 25 = rhs)
  float* xq     = (float*)(smem + 28800);          // [25]
  float* ksv    = (float*)(smem + 29056);          // [125]
  float* Sbuf   = (float*)(smem + 40960);          // [32][128] f32 scores
  float* statf  = (float*)(smem + 57344);          // stats scratch (512 floats)
  float* ssq    = statf;                           // [32]
  float* qscl   = statf + 32;                      // [32]
  float* rowmax = statf + 64;                      // [25]
  float* rowsum = statf + 96;                      // [25]
  float* colmax = statf + 128;                     // [125]
  float* colsum = statf + 256;                     // [125]
  float* misc   = statf + 384;                     // pinv, inv sums

  const float* qbase = qry + (size_t)(b * QCNT + qi) * CDIM * HW;  // [c][p]

  // ---- stage 1: per-position sum of squares over c ----
  if (tid < 32) ssq[tid] = 0.0f;
  __syncthreads();
  for (int idx = tid; idx < CDIM * HW; idx += 256) {
    float v = qbase[idx];
    atomicAdd(&ssq[idx % HW], v * v);
  }
  __syncthreads();
  if (tid < 32) qscl[tid] = rsqrtf(ssq[tid] + 1e-12f);
  __syncthreads();

  // ---- stage 2: normalized f16 query, A-layout Qh[i][c]; pad rows 25..31 ----
  for (int idx = tid; idx < CDIM * HW; idx += 256) {
    int c = idx / HW, p = idx % HW;
    Qh[p * CDIM + c] = (_Float16)(qbase[idx] * qscl[p]);
  }
  for (int idx = tid; idx < (32 - HW) * CDIM; idx += 256)
    Qh[HW * CDIM + idx] = (_Float16)0.0f;
  __syncthreads();

  // ---- stage 3: S = Q * Sh^T via v_wmma_f32_16x16x32_f16 ----
  // wave -> (itile = wave>>2, jtiles = (wave&3)*2, +1); A fragment shared.
  {
    const int itile = wave >> 2;
    const int jt0   = (wave & 3) * 2;
    const int mrow  = lane & 15;
    const int khalf = lane >> 4;
    const _Float16* qrow  = Qh + (itile * 16 + mrow) * CDIM + khalf * 16;
    const _Float16* sbase = Sh + (size_t)b * JPAD * CDIM;
    const _Float16* brow0 = sbase + (size_t)(jt0 * 16 + mrow) * CDIM + khalf * 16;
    const _Float16* brow1 = brow0 + (size_t)16 * CDIM;

    v8f acc0 = {}; v8f acc1 = {};
    for (int kk = 0; kk < CDIM / 32; ++kk) {
      const int c0 = kk * 32;
      v8h alo = *(const v8h*)(qrow + c0);
      v8h ahi = *(const v8h*)(qrow + c0 + 8);
      v16h a;
#pragma unroll
      for (int e = 0; e < 8; ++e) { a[e] = alo[e]; a[e + 8] = ahi[e]; }
      v8h b0l = *(const v8h*)(brow0 + c0);
      v8h b0h = *(const v8h*)(brow0 + c0 + 8);
      v16h bf0;
#pragma unroll
      for (int e = 0; e < 8; ++e) { bf0[e] = b0l[e]; bf0[e + 8] = b0h[e]; }
      v8h b1l = *(const v8h*)(brow1 + c0);
      v8h b1h = *(const v8h*)(brow1 + c0 + 8);
      v16h bf1;
#pragma unroll
      for (int e = 0; e < 8; ++e) { bf1[e] = b1l[e]; bf1[e + 8] = b1h[e]; }

      acc0 = __builtin_amdgcn_wmma_f32_16x16x32_f16(false, a, false, bf0,
                                                    (short)0, acc0, false, false);
      acc1 = __builtin_amdgcn_wmma_f32_16x16x32_f16(false, a, false, bf1,
                                                    (short)0, acc1, false, false);
    }
    // D layout: VGPR v -> row itile*16 + v + 8*(lane>>4), col = jtile*16 + (lane&15)
    const int col0  = jt0 * 16 + (lane & 15);
    const int rbase = itile * 16 + (lane >> 4) * 8;
#pragma unroll
    for (int v = 0; v < 8; ++v) {
      Sbuf[(rbase + v) * JPAD + col0]      = acc0[v];
      Sbuf[(rbase + v) * JPAD + col0 + 16] = acc1[v];
    }
  }
  __syncthreads();   // Qh dead after this point; Tsq/Tqs may reuse its space

  // ---- stage 4a: row stats (softmax over j, gamma=20) ----
  for (int r = wave; r < HW; r += 8) {
    float mx = -3.0e38f;
    for (int j = lane; j < MS; j += 32) mx = fmaxf(mx, Sbuf[r * JPAD + j]);
    mx = wave_max(mx);
    float sm = 0.0f;
    for (int j = lane; j < MS; j += 32) sm += __expf(GAMMA1 * (Sbuf[r * JPAD + j] - mx));
    sm = wave_sum(sm);
    if (lane == 0) { rowmax[r] = mx; rowsum[r] = sm; }
  }
  // ---- stage 4b: column stats (softmax over i, gamma=10) ----
  if (tid < MS) {
    float mx = -3.0e38f;
    for (int i = 0; i < HW; ++i) mx = fmaxf(mx, Sbuf[i * JPAD + tid]);
    float sm = 0.0f;
    for (int i = 0; i < HW; ++i) sm += __expf(GAMMA2 * (Sbuf[i * JPAD + tid] - mx));
    colmax[tid] = mx; colsum[tid] = sm;
  }
  __syncthreads();

  // ---- stage 4c: materialize Tsq[i][j], Tqs[i][j] (= T_qs[j][i]) ----
  for (int idx = tid; idx < HW * JPAD; idx += 256) {
    const int i = idx >> 7, j = idx & 127;
    if (j < MS) {
      const float s = Sbuf[i * JPAD + j];
      Tsq[idx] = __expf(GAMMA1 * (s - rowmax[i])) / rowsum[i];
      Tqs[idx] = __expf(GAMMA2 * (s - colmax[j])) / colsum[j];
    } else { Tsq[idx] = 0.0f; Tqs[idx] = 0.0f; }
  }
  __syncthreads();

  // ---- stage 4d: Mx = I - c^2 * (B*A); rhs (col 25) = 1 + c * B*1 ----
  const float c2 = KF * KF;
  for (int idx = tid; idx < HW * HW; idx += 256) {
    const int i = idx / HW, ip = idx % HW;
    float g = 0.0f;
    for (int j = 0; j < MS; ++j) g += Tqs[i * JPAD + j] * Tsq[ip * JPAD + j];
    Mx[i * 32 + ip] = ((i == ip) ? 1.0f : 0.0f) - c2 * g;
  }
  if (tid < HW) {
    float s = 0.0f;
    for (int j = 0; j < MS; ++j) s += Tqs[tid * JPAD + j];
    Mx[tid * 32 + HW] = 1.0f + KF * s;
  }
  __syncthreads();

  // ---- stage 4e: pivot-free Gaussian elimination (column diag dominant) ----
  for (int k = 0; k < HW; ++k) {
    if (tid == 0) misc[0] = 1.0f / Mx[k * 32 + k];
    __syncthreads();
    const float pinv = misc[0];
    for (int idx = tid; idx < HW * (HW + 1); idx += 256) {
      const int i = idx / (HW + 1), j = idx % (HW + 1);  // j==25 is rhs
      if (i > k && j > k)
        Mx[i * 32 + j] -= Mx[i * 32 + k] * pinv * Mx[k * 32 + j];
    }
    __syncthreads();
  }
  // back substitution
  for (int k = HW - 1; k >= 0; --k) {
    if (tid == 0) xq[k] = Mx[k * 32 + HW] / Mx[k * 32 + k];
    __syncthreads();
    if (tid < k) Mx[tid * 32 + HW] -= Mx[tid * 32 + k] * xq[k];
    __syncthreads();
  }

  // ---- stage 5: ks = c * A * xq ; output normalizations ----
  if (tid < MS) {
    float a = 0.0f;
    for (int i = 0; i < HW; ++i) a += Tsq[i * JPAD + tid] * xq[i];
    ksv[tid] = KF * a;
  }
  __syncthreads();
  if (tid == 0) {
    float s = 0.0f;
    for (int i = 0; i < HW; ++i) s += xq[i] - 1.0f;
    misc[1] = 1.0f / s;
  }
  if (tid < NWAY) {
    float s = 0.0f;
    for (int p = 0; p < HW; ++p) s += ksv[tid * HW + p];
    misc[2 + tid] = 1.0f / s;
  }
  __syncthreads();
  if (tid < HW)
    out_q[(size_t)inst * HW + tid] = (xq[tid] - 1.0f) * misc[1];
  if (tid < MS)
    out_s[(size_t)inst * MS + tid] = ksv[tid] * misc[2 + tid / HW];
}

// ---------------------------------------------------------------------------
extern "C" void kernel_launch(void* const* d_in, const int* in_sizes, int n_in,
                              void* d_out, int out_size, void* d_ws, size_t ws_size,
                              hipStream_t stream) {
  const float* sup = (const float*)d_in[0];
  const float* qry = (const float*)d_in[1];
  // d_in[2]=n_way, d_in[3]=k_shot are compile-time constants (5,5)

  _Float16* Sh = (_Float16*)d_ws;   // 32*128*640 f16 = 5.24 MB

  k_prep_support<<<dim3(BATCH * NWAY), dim3(256), 0, stream>>>(sup, Sh);

  float* out_q = (float*)d_out;                  // (b,q,1,h,w)  = 60000
  float* out_s = out_q + BATCH * QCNT * HW;      // (b,q,n,1,h,w) = 300000
  k_katz<<<dim3(BATCH * QCNT), dim3(256), 0, stream>>>(qry, Sh, out_q, out_s);
}
